// CAN_Smiles_Layer_20452634263707
// MI455X (gfx1250) — matrix operand order repaired and checked by reading, not compiled, not run on hardware
//
#include <hip/hip_runtime.h>
#include <hip/hip_bf16.h>

typedef __bf16 bf16;
typedef __attribute__((ext_vector_type(16))) __bf16 v16bf;
typedef __attribute__((ext_vector_type(8)))  __bf16 v8bf;
typedef __attribute__((ext_vector_type(8)))  float  v8f;

#define WMMA_BF16(A, B, C) \
  __builtin_amdgcn_wmma_f32_16x16x32_bf16(false, (A), false, (B), (short)0, (C), false, false)

// ---- gfx1250 async-to-LDS support (guarded; falls back to direct global) ----
#if defined(__gfx1250__) && __has_builtin(__builtin_amdgcn_global_load_async_to_lds_b128)
#define HAVE_ASYNC 1
#else
#define HAVE_ASYNC 0
#endif

#if HAVE_ASYNC
#if __has_builtin(__builtin_amdgcn_s_wait_asynccnt)
#define WAIT_ASYNC(n) __builtin_amdgcn_s_wait_asynccnt(n)
#else
#define WAIT_ASYNC(n) asm volatile("s_wait_asynccnt %0" ::"i"(n) : "memory")
#endif
typedef int v4i __attribute__((ext_vector_type(4)));
typedef __attribute__((address_space(1))) v4i* g_v4i_p;  // global int4*
typedef __attribute__((address_space(3))) v4i* l_v4i_p;  // LDS int4*
#endif

// ---------------------------------------------------------------------------
// Stage 1: group-mean (4 consecutive rows) + f32 -> bf16 convert. D = 256.
// ---------------------------------------------------------------------------
__global__ void group_mean_kernel(const float* __restrict__ src, bf16* __restrict__ dst,
                                  int Lg, long total) {
  long idx = (long)blockIdx.x * blockDim.x + threadIdx.x;
  if (idx >= total) return;
  int ch = (int)(idx & 255);
  long t = idx >> 8;
  int g = (int)(t % Lg);
  int b = (int)(t / Lg);
  const float* pr = src + ((size_t)b * Lg * 4 + (size_t)g * 4) * 256 + ch;
  float s = pr[0] + pr[256] + pr[512] + pr[768];
  dst[idx] = (bf16)(s * 0.25f);
}

// Grouped mask (any over 4) bit-packed into u32 words (32 groups per word).
__global__ void pack_mask_kernel(const int* __restrict__ mask, unsigned* __restrict__ bits,
                                 int Lg, int wordsPerB, int totalWords) {
  int idx = blockIdx.x * blockDim.x + threadIdx.x;
  if (idx >= totalWords) return;
  int b = idx / wordsPerB;
  int w = idx % wordsPerB;
  unsigned word = 0;
  for (int j = 0; j < 32; ++j) {
    int g = w * 32 + j;
    const int* mp = mask + (size_t)b * Lg * 4 + (size_t)g * 4;
    int any = mp[0] | mp[1] | mp[2] | mp[3];
    word |= (any ? 1u : 0u) << j;
  }
  bits[idx] = word;
}

// Wt[c][f] = (bf16)W[f][c]  (column-major weights -> contiguous B-fragment rows)
__global__ void weight_transpose_kernel(const float* __restrict__ W, bf16* __restrict__ Wt) {
  int idx = blockIdx.x * blockDim.x + threadIdx.x;  // 65536 total
  int c = idx >> 8, f = idx & 255;
  Wt[idx] = (bf16)W[(size_t)f * 256 + c];
}

// ---------------------------------------------------------------------------
// Stage 2: projection GEMM  Out[rows x 256] = X[rows x 256] * W[256 x 256]
// One wave per 16x64 output tile; 8 K-chunks of 32; WMMA bf16 -> f32 acc.
// transposed=1 stores Out as [chan][row] (for V matrices) with b128 stores.
// ---------------------------------------------------------------------------
__global__ __launch_bounds__(32) void proj_gemm_kernel(
    const bf16* __restrict__ X, int rows, const bf16* __restrict__ Wt,
    bf16* __restrict__ Out, int transposed) {
  int lane = threadIdx.x;
  int r = lane & 15, half = lane >> 4, hsel = half * 8;
  int b = blockIdx.z;
  int mTile = blockIdx.x;
  int n0 = blockIdx.y * 64;

  const bf16* Xrow = X + ((size_t)b * rows + mTile * 16 + r) * 256;
  v8f z{};
  v8f acc[4];
#pragma unroll
  for (int t = 0; t < 4; ++t) acc[t] = z;

#pragma unroll
  for (int kc = 0; kc < 8; ++kc) {
    int k0 = kc * 32;
    // A fragment: row = token, two runs of 8 consecutive features
    v8bf lo = *(const v8bf*)(Xrow + k0 + half * 16);
    v8bf hi = *(const v8bf*)(Xrow + k0 + half * 16 + 8);
    v16bf a;
#pragma unroll
    for (int j = 0; j < 8; ++j) { a[j] = lo[j]; a[8 + j] = hi[j]; }
#pragma unroll
    for (int t = 0; t < 4; ++t) {
      // B fragment: column = out-channel, 16 consecutive features from Wt row
      const bf16* wp = Wt + (size_t)(n0 + t * 16 + r) * 256 + k0 + half * 16;
      v16bf bfrag = *(const v16bf*)wp;
      acc[t] = WMMA_BF16(a, bfrag, acc[t]);
    }
  }
  if (transposed) {
#pragma unroll
    for (int t = 0; t < 4; ++t) {
      int chan = n0 + t * 16 + r;
      v8bf pk;
#pragma unroll
      for (int i = 0; i < 8; ++i) pk[i] = (bf16)acc[t][i];
      // 8 consecutive rows -> one 16B store
      *(v8bf*)(Out + ((size_t)b * 256 + chan) * rows + mTile * 16 + hsel) = pk;
    }
  } else {
#pragma unroll
    for (int t = 0; t < 4; ++t) {
      int chan = n0 + t * 16 + r;
#pragma unroll
      for (int i = 0; i < 8; ++i) {
        int row = mTile * 16 + i + hsel;
        Out[((size_t)b * rows + row) * 256 + chan] = (bf16)acc[t][i];
      }
    }
  }
}

// ---------------------------------------------------------------------------
// Stage 3: attention. Block = 8 waves = 8 heads, one (batch, 16-query tile).
// All heads share the same 32 key rows per chunk -> stage the flat 16KB
// K tile in LDS via double-buffered async copies (8x traffic reduction).
// S^T = K*Q^T (key-major fragment => cheap per-query softmax); exp(S^T)
// converts to the A(P) fragment with no cross-lane movement; O += P*V^T.
// ---------------------------------------------------------------------------
#define STAGE_ELEMS (32 * 256)  // one chunk: 32 keys x 256 chans (bf16) = 16KB

#if HAVE_ASYNC
__device__ __forceinline__ void issue_copy_async(const bf16* Kb, int keyBase,
                                                 bf16* lbuf, int tid) {
  // Flat contiguous 16KB tile; each of 256 threads copies 64B = 4 x b128.
  char* src = (char*)(Kb + (size_t)keyBase * 256) + tid * 64;
  char* dst = (char*)lbuf + tid * 64;
#pragma unroll
  for (int j = 0; j < 4; ++j) {
    __builtin_amdgcn_global_load_async_to_lds_b128(
        (g_v4i_p)(src + j * 16), (l_v4i_p)(dst + j * 16), 0, 0);
  }
}
#endif

__device__ __forceinline__ void attn_pass(
    const bf16* __restrict__ Kb,    // K matrix for batch b (no head offset), row stride 256
    int hoff,                       // h * 32 column offset
    const bf16* __restrict__ Vtbh,  // V^T for (b), + h*32 row offset; row stride Ltok
    int Ltok, const unsigned* __restrict__ colBits,
    v16bf bq, int r, int half, int tid, bf16* stage,
    v8f& o0, v8f& o1, float& lsum) {
  float mrun = -3.0e38f, lrun = 0.0f;
  v8f z{};
  o0 = z; o1 = z;
  int hsel = half * 8;
  int nChunks = Ltok >> 5;

#if HAVE_ASYNC
  issue_copy_async(Kb, 0, stage, tid);
#endif

  for (int c = 0; c < nChunks; ++c) {
    int keyBase = c * 32;
    unsigned mb = colBits[c];

#if HAVE_ASYNC
    if (c + 1 < nChunks)
      issue_copy_async(Kb, keyBase + 32, stage + ((c + 1) & 1) * STAGE_ELEMS, tid);
    if (c + 1 < nChunks) { WAIT_ASYNC(4); } else { WAIT_ASYNC(0); }
    __syncthreads();  // chunk c fully staged by all waves
    const bf16* rp0 = stage + (size_t)((c & 1) * STAGE_ELEMS) + (size_t)r * 256 + hoff;
    const bf16* rp1 = rp0 + 16 * 256;
#else
    const bf16* rp0 = Kb + (size_t)(keyBase + r) * 256 + hoff;
    const bf16* rp1 = rp0 + 16 * 256;
    if (c + 1 < nChunks) __builtin_prefetch(rp0 + 32 * 256, 0, 3);
#endif

    // A fragments: K tiles (16 keys x 32 d each)
    v8bf a0lo = *(const v8bf*)(rp0 + half * 16);
    v8bf a0hi = *(const v8bf*)(rp0 + half * 16 + 8);
    v8bf a1lo = *(const v8bf*)(rp1 + half * 16);
    v8bf a1hi = *(const v8bf*)(rp1 + half * 16 + 8);
    v16bf a0, a1;
#pragma unroll
    for (int j = 0; j < 8; ++j) {
      a0[j] = a0lo[j]; a0[8 + j] = a0hi[j];
      a1[j] = a1lo[j]; a1[8 + j] = a1hi[j];
    }
    v8f s0 = WMMA_BF16(a0, bq, z);  // S^T: vgpr i = key (hsel+i), lane%16 = query
    v8f s1 = WMMA_BF16(a1, bq, z);  //      keys 16 + hsel + i

    float t0[8], t1[8], cmax = -3.0e38f;
#pragma unroll
    for (int i = 0; i < 8; ++i) {
      float v = s0[i] + (((mb >> (hsel + i)) & 1u) ? 0.0f : -1.0e6f);
      t0[i] = v; cmax = fmaxf(cmax, v);
    }
#pragma unroll
    for (int i = 0; i < 8; ++i) {
      float v = s1[i] + (((mb >> (16 + hsel + i)) & 1u) ? 0.0f : -1.0e6f);
      t1[i] = v; cmax = fmaxf(cmax, v);
    }
    cmax = fmaxf(cmax, __shfl_xor(cmax, 16));
    float mnew = fmaxf(mrun, cmax);
    float corr = __expf(mrun - mnew);
    float psum = 0.0f;
    v16bf ap;  // A fragment of P: j<8 <-> keys hsel+j, j>=8 <-> 16+hsel+(j-8)
#pragma unroll
    for (int i = 0; i < 8; ++i) { float e = __expf(t0[i] - mnew); psum += e; ap[i] = (bf16)e; }
#pragma unroll
    for (int i = 0; i < 8; ++i) { float e = __expf(t1[i] - mnew); psum += e; ap[8 + i] = (bf16)e; }
    psum += __shfl_xor(psum, 16);
    lrun = lrun * corr + psum;
    mrun = mnew;

    // Rescale O accumulators (O vgpr i belongs to query hsel+i; stats in lanes q, q+16)
#pragma unroll
    for (int i = 0; i < 8; ++i) {
      float cq = __shfl(corr, hsel + i);
      o0[i] *= cq; o1[i] *= cq;
    }
    // B fragments from V^T: row = d-channel, keys hsel+j and 16+hsel+j
    const bf16* vp0 = Vtbh + (size_t)r * Ltok + keyBase + hsel;
    const bf16* vp1 = Vtbh + (size_t)(16 + r) * Ltok + keyBase + hsel;
    v8bf bv0lo = *(const v8bf*)vp0, bv0hi = *(const v8bf*)(vp0 + 16);
    v8bf bv1lo = *(const v8bf*)vp1, bv1hi = *(const v8bf*)(vp1 + 16);
    v16bf bv0, bv1;
#pragma unroll
    for (int j = 0; j < 8; ++j) {
      bv0[j] = bv0lo[j]; bv0[8 + j] = bv0hi[j];
      bv1[j] = bv1lo[j]; bv1[8 + j] = bv1hi[j];
    }
    o0 = WMMA_BF16(ap, bv0, o0);
    o1 = WMMA_BF16(ap, bv1, o1);

#if HAVE_ASYNC
    __syncthreads();  // all waves done reading buf[c&1] before it is reused
#endif
  }
  lsum = lrun;
}

__global__ __launch_bounds__(256) void attn_kernel(
    const bf16* __restrict__ Kp, const bf16* __restrict__ Vtp,
    const bf16* __restrict__ Qs, const bf16* __restrict__ Ks,
    const bf16* __restrict__ Vts,
    const unsigned* __restrict__ mprotBits, const unsigned* __restrict__ msmBits,
    float* __restrict__ out) {
  __shared__ bf16 kstage[2 * STAGE_ELEMS];  // 32KB double buffer
  int tid = threadIdx.x;
  int lane = tid & 31;
  int h = tid >> 5;  // wave == head
  int r = lane & 15, half = lane >> 4, hsel = half * 8;
  int qTile = blockIdx.x;  // 16 tiles of 16 queries
  int b = blockIdx.y;

  // B(Q^T) fragment: column = query, 16 consecutive d values (head dim = 32 = WMMA K)
  int qrow = qTile * 16 + r;
  const bf16* qp = Qs + ((size_t)b * 256 + qrow) * 256 + h * 32 + half * 16;
  v16bf bq = *(const v16bf*)qp;

  const bf16* Kp_b   = Kp  + (size_t)b * 1024 * 256;
  const bf16* Vtp_bh = Vtp + ((size_t)b * 256 + h * 32) * 1024;
  const bf16* Ks_b   = Ks  + (size_t)b * 256 * 256;
  const bf16* Vts_bh = Vts + ((size_t)b * 256 + h * 32) * 256;

  v8f odp0, odp1, odd0, odd1;
  float ldp, ldd;
  attn_pass(Kp_b, h * 32, Vtp_bh, 1024, mprotBits + b * 32, bq, r, half, tid, kstage,
            odp0, odp1, ldp);
  attn_pass(Ks_b, h * 32, Vts_bh, 256, msmBits + b * 8, bq, r, half, tid, kstage,
            odd0, odd1, ldd);

  float invdp = 1.0f / ldp;
  float invdd = 1.0f / ldd;
  unsigned mw = msmBits[b * 8 + (qTile >> 1)];
  float mrowF = ((mw >> ((qTile & 1) * 16 + r)) & 1u) ? 1.0f : 0.0f;

  float* outb = out + (size_t)b * 256 * 256;
#pragma unroll
  for (int i = 0; i < 8; ++i) {
    int src = hsel + i;
    float s1 = __shfl(invdp, src);
    float s2 = __shfl(invdd, src);
    float mr = __shfl(mrowF, src);
    int row = qTile * 16 + i + hsel;
    float v0 = (odp0[i] * s1 + odd0[i] * s2) * 0.5f * mr;
    float v1 = (odp1[i] * s1 + odd1[i] * s2) * 0.5f * mr;
    outb[(size_t)row * 256 + h * 32 + r] = v0;
    outb[(size_t)row * 256 + h * 32 + 16 + r] = v1;
  }
}

// ---------------------------------------------------------------------------
extern "C" void kernel_launch(void* const* d_in, const int* in_sizes, int n_in,
                              void* d_out, int out_size, void* d_ws, size_t ws_size,
                              hipStream_t stream) {
  (void)in_sizes; (void)n_in; (void)out_size; (void)ws_size;
  const float* protein   = (const float*)d_in[0];
  const float* sm        = (const float*)d_in[1];
  const int*   mask_prot = (const int*)d_in[2];
  const int*   mask_sm   = (const int*)d_in[3];
  const float* Wk_p = (const float*)d_in[4];
  const float* Wv_p = (const float*)d_in[5];
  const float* Wq_d = (const float*)d_in[6];
  const float* Wk_d = (const float*)d_in[7];
  const float* Wv_d = (const float*)d_in[8];
  float* out = (float*)d_out;

  char* p = (char*)d_ws;
  auto carve = [&](size_t bytes) -> char* {
    char* r = p;
    p += (bytes + 255) & ~(size_t)255;
    return r;
  };
  const size_t NPg = 32ull * 1024 * 256;  // grouped protein elems
  const size_t NSg = 32ull * 256 * 256;   // grouped sm elems
  bf16* protg = (bf16*)carve(NPg * 2);
  bf16* smg   = (bf16*)carve(NSg * 2);
  bf16* WtKp  = (bf16*)carve(65536ull * 2);
  bf16* WtVp  = (bf16*)carve(65536ull * 2);
  bf16* WtQd  = (bf16*)carve(65536ull * 2);
  bf16* WtKd  = (bf16*)carve(65536ull * 2);
  bf16* WtVd  = (bf16*)carve(65536ull * 2);
  bf16* Kp    = (bf16*)carve(NPg * 2);
  bf16* Vtp   = (bf16*)carve(NPg * 2);   // transposed [b][chan][1024]
  bf16* Qsb   = (bf16*)carve(NSg * 2);
  bf16* Ksb   = (bf16*)carve(NSg * 2);
  bf16* Vts   = (bf16*)carve(NSg * 2);   // transposed [b][chan][256]
  unsigned* mprotBits = (unsigned*)carve(1024 * 4);  // 32 b x 32 words
  unsigned* msmBits   = (unsigned*)carve(256 * 4);   // 32 b x 8 words

  group_mean_kernel<<<(unsigned)(NPg / 256), 256, 0, stream>>>(protein, protg, 1024, (long)NPg);
  group_mean_kernel<<<(unsigned)(NSg / 256), 256, 0, stream>>>(sm, smg, 256, (long)NSg);
  pack_mask_kernel<<<4, 256, 0, stream>>>(mask_prot, mprotBits, 1024, 32, 1024);
  pack_mask_kernel<<<1, 256, 0, stream>>>(mask_sm, msmBits, 256, 8, 256);
  weight_transpose_kernel<<<256, 256, 0, stream>>>(Wk_p, WtKp);
  weight_transpose_kernel<<<256, 256, 0, stream>>>(Wv_p, WtVp);
  weight_transpose_kernel<<<256, 256, 0, stream>>>(Wq_d, WtQd);
  weight_transpose_kernel<<<256, 256, 0, stream>>>(Wk_d, WtKd);
  weight_transpose_kernel<<<256, 256, 0, stream>>>(Wv_d, WtVd);

  proj_gemm_kernel<<<dim3(64, 4, 32), 32, 0, stream>>>(protg, 1024, WtKp, Kp, 0);
  proj_gemm_kernel<<<dim3(64, 4, 32), 32, 0, stream>>>(protg, 1024, WtVp, Vtp, 1);
  proj_gemm_kernel<<<dim3(16, 4, 32), 32, 0, stream>>>(smg, 256, WtQd, Qsb, 0);
  proj_gemm_kernel<<<dim3(16, 4, 32), 32, 0, stream>>>(smg, 256, WtKd, Ksb, 0);
  proj_gemm_kernel<<<dim3(16, 4, 32), 32, 0, stream>>>(smg, 256, WtVd, Vts, 1);

  attn_kernel<<<dim3(16, 32), 256, 0, stream>>>(Kp, Vtp, Qsb, Ksb, Vts,
                                                mprotBits, msmBits, out);
}
